// PETDecoder_12034498363963
// MI455X (gfx1250) — matrix operand order repaired and checked by reading, not compiled, not run on hardware
//
#include <hip/hip_runtime.h>
#include <hip/hip_bf16.h>
#include <math.h>

typedef __attribute__((ext_vector_type(16))) _Float16 v16h;
typedef __attribute__((ext_vector_type(8)))  _Float16 v8h;
typedef __attribute__((ext_vector_type(8)))  float    v8f;

#define BATCH 8
#define CDIM  256
#define KCONV 512
#define HF    128
#define HWF   16384
#define NQ    14745

#define PITCH 40   // LDS row pitch in halfs (80B = 5x16B, keeps 16B alignment of sub-runs)
#define PITCH32 (PITCH / 2)

union PackU { _Float16 h[2]; unsigned u; };

// ---------------------------------------------------------------------------
// WMMA fragment helpers (CDNA5 ISA 7.12.2 16-bit layouts, wave32)
// A 16x32: lanes 0-15 -> M=lane, K={0..7,16..23}; lanes 16-31 -> M=lane-16, K={8..15,24..31}
// B 32x16: lanes 0-15 -> N=lane, K={0..7,16..23}; lanes 16-31 -> N=lane-16, K={8..15,24..31}
// LDS holds A as [m][k] (k-contig) and B transposed as [n][k] (k-contig), so each
// fragment is two 16-byte contiguous runs -> 2x ds_load_b128 per fragment.
// ---------------------------------------------------------------------------
__device__ __forceinline__ v16h cat8(v8h a, v8h b) {
  return __builtin_shufflevector(a, b, 0, 1, 2, 3, 4, 5, 6, 7,
                                 8, 9, 10, 11, 12, 13, 14, 15);
}

__device__ __forceinline__ v16h load_a_frag(const _Float16* As, int mo, int lane) {
  const _Float16* base = As + (mo + (lane & 15)) * PITCH + ((lane < 16) ? 0 : 8);
  v8h x = *(const v8h*)(base);
  v8h y = *(const v8h*)(base + 16);
  return cat8(x, y);
}

__device__ __forceinline__ v16h load_b_frag(const _Float16* Bs, int ncol, int lane) {
  const _Float16* base = Bs + (ncol + (lane & 15)) * PITCH + ((lane < 16) ? 0 : 8);
  v8h x = *(const v8h*)(base);
  v8h y = *(const v8h*)(base + 16);
  return cat8(x, y);
}

// ---------------------------------------------------------------------------
// GEMM 1: mem[b,o,n] = sum_c conv_w[o,c]*X[b,c,n] + conv_b[o]
// X[c,n] = c<256 ? upsample2x(enc)[c,n] : feat4x[c-256,n]
// M=256 (o), N=16384 (n), K=512. Block tile 128x64, 4 waves, wave strip 32x64.
// ---------------------------------------------------------------------------
__global__ void __launch_bounds__(128, 1)
k_gemm_mem(const float* __restrict__ enc, const float* __restrict__ f4,
           const float* __restrict__ W, const float* __restrict__ bias,
           float* __restrict__ out) {
  __shared__ _Float16 As[128 * PITCH];
  __shared__ _Float16 Bs[64 * PITCH];
  int b = blockIdx.z;
  int mbase = blockIdx.y * 128;
  int nbase = blockIdx.x * 64;
  int tid = threadIdx.x;
  int wv = tid >> 5, lane = tid & 31;
  int mo = wv * 32;
  bool lo = lane < 16;

  int r0 = tid >> 4;        // 0..7
  int kp = tid & 15;        // fixed k-pair
  unsigned* AsU = (unsigned*)As;
  unsigned* BsU = (unsigned*)Bs;
  int sIdx = r0 * PITCH32 + kp;

  // invariant bases (int offsets; all extents < 2^31)
  int aoff = (mbase + r0) * KCONV + 2 * kp;                 // conv_w
  const float* encB = enc + (b * CDIM) * 4096;              // + k*4096 + spatial
  const float* f4B = f4 + (b * CDIM) * HWF;                 // + (k-256)*HWF + spatial
  // spatial offsets for this thread's 8 B rows (fixed across K)
  int eoff[8], foff[8];
#pragma unroll
  for (int t = 0; t < 8; ++t) {
    int ng = nbase + r0 + 8 * t;
    int y = ng >> 7, x = ng & 127;
    eoff[t] = (y >> 1) * 64 + (x >> 1);
    foff[t] = y * HF + x;
  }

  v8f acc[2][4];
#pragma unroll
  for (int s = 0; s < 2; ++s)
#pragma unroll
    for (int t = 0; t < 4; ++t)
#pragma unroll
      for (int r = 0; r < 8; ++r) acc[s][t][r] = 0.f;

  for (int kk = 0; kk < KCONV; kk += 32) {
    // A tile: 128x32 conv_w
#pragma unroll
    for (int t = 0; t < 16; ++t) {
      const float* ap = W + aoff + t * 8 * KCONV;
      PackU u;
      u.h[0] = (_Float16)ap[0];
      u.h[1] = (_Float16)ap[1];
      AsU[sIdx + t * 8 * PITCH32] = u.u;
    }
    // B tile (transposed in LDS): 64 n x 32 k
    int k0 = kk + 2 * kp;
    if (k0 < CDIM) {
      const float* e = encB + k0 * 4096;
#pragma unroll
      for (int t = 0; t < 8; ++t) {
        PackU u;
        u.h[0] = (_Float16)e[eoff[t]];
        u.h[1] = (_Float16)e[eoff[t] + 4096];
        BsU[sIdx + t * 8 * PITCH32] = u.u;
      }
    } else {
      const float* f = f4B + (k0 - CDIM) * HWF;
#pragma unroll
      for (int t = 0; t < 8; ++t) {
        PackU u;
        u.h[0] = (_Float16)f[foff[t]];
        u.h[1] = (_Float16)f[foff[t] + HWF];
        BsU[sIdx + t * 8 * PITCH32] = u.u;
      }
    }
    __syncthreads();
    v16h bf0 = load_b_frag(Bs, 0, lane);
    v16h bf1 = load_b_frag(Bs, 16, lane);
    v16h bf2 = load_b_frag(Bs, 32, lane);
    v16h bf3 = load_b_frag(Bs, 48, lane);
#pragma unroll
    for (int s = 0; s < 2; ++s) {
      v16h af = load_a_frag(As, mo + 16 * s, lane);
      acc[s][0] = __builtin_amdgcn_wmma_f32_16x16x32_f16(false, af, false, bf0, (short)0, acc[s][0], false, false);
      acc[s][1] = __builtin_amdgcn_wmma_f32_16x16x32_f16(false, af, false, bf1, (short)0, acc[s][1], false, false);
      acc[s][2] = __builtin_amdgcn_wmma_f32_16x16x32_f16(false, af, false, bf2, (short)0, acc[s][2], false, false);
      acc[s][3] = __builtin_amdgcn_wmma_f32_16x16x32_f16(false, af, false, bf3, (short)0, acc[s][3], false, false);
    }
    __syncthreads();
    aoff += 32;
  }
  int obase = b * CDIM * HWF;
#pragma unroll
  for (int s = 0; s < 2; ++s)
#pragma unroll
    for (int t = 0; t < 4; ++t) {
      int n = nbase + t * 16 + (lane & 15);
#pragma unroll
      for (int r = 0; r < 8; ++r) {
        int m = mbase + mo + 16 * s + (lo ? r : 8 + r);
        out[obase + m * HWF + n] = acc[s][t][r] + bias[m];
      }
    }
}

// ---------------------------------------------------------------------------
// Generic GEMM: C[b,m,n] = act( sum_k A[b,m,k]*W[n,k] + bias[n] )
// Int strides (all extents < 2^31), optional drop mask on m, W row-major [N,K],
// C row-major [M,N]. N,K multiples of 64/32; M guarded. Block tile 128x64.
// ---------------------------------------------------------------------------
__global__ void __launch_bounds__(128, 1)
k_gemm(const float* __restrict__ A, int a_b, int a_m, int a_k,
       const unsigned char* __restrict__ dropp, int drop_b,
       const float* __restrict__ W, const float* __restrict__ bias,
       float* __restrict__ C, int c_b,
       int M, int N, int K, int relu) {
  __shared__ _Float16 As[128 * PITCH];
  __shared__ _Float16 Bs[64 * PITCH];
  int b = blockIdx.z;
  int mbase = blockIdx.y * 128;
  int nbase = blockIdx.x * 64;
  int tid = threadIdx.x;
  int wv = tid >> 5, lane = tid & 31;
  int mo = wv * 32;
  bool lo = lane < 16;

  int r0 = tid >> 4;
  int kp = tid & 15;
  unsigned* AsU = (unsigned*)As;
  unsigned* BsU = (unsigned*)Bs;
  int sIdx = r0 * PITCH32 + kp;

  // zero mask for this thread's 16 A rows (fixed across K): OOB or dropped
  unsigned zmask = 0;
#pragma unroll
  for (int t = 0; t < 16; ++t) {
    int m = mbase + r0 + 8 * t;
    bool z = (m >= M) || (dropp && m < M && dropp[b * drop_b + m]);
    zmask |= (z ? 1u : 0u) << t;
  }
  int aoff = b * a_b + (mbase + r0) * a_m + 2 * kp * a_k;
  int woff = (nbase + r0) * K + 2 * kp;

  v8f acc[2][4];
#pragma unroll
  for (int s = 0; s < 2; ++s)
#pragma unroll
    for (int t = 0; t < 4; ++t)
#pragma unroll
      for (int r = 0; r < 8; ++r) acc[s][t][r] = 0.f;

  for (int kk = 0; kk < K; kk += 32) {
#pragma unroll
    for (int t = 0; t < 16; ++t) {
      float v0 = 0.f, v1 = 0.f;
      if (!((zmask >> t) & 1u)) {
        const float* ap = A + aoff + t * 8 * a_m;
        v0 = ap[0];
        v1 = ap[a_k];
      }
      PackU u;
      u.h[0] = (_Float16)v0;
      u.h[1] = (_Float16)v1;
      AsU[sIdx + t * 8 * PITCH32] = u.u;
    }
#pragma unroll
    for (int t = 0; t < 8; ++t) {
      const float* wp = W + woff + t * 8 * K;
      PackU u;
      u.h[0] = (_Float16)wp[0];
      u.h[1] = (_Float16)wp[1];
      BsU[sIdx + t * 8 * PITCH32] = u.u;
    }
    __syncthreads();
    v16h bf0 = load_b_frag(Bs, 0, lane);
    v16h bf1 = load_b_frag(Bs, 16, lane);
    v16h bf2 = load_b_frag(Bs, 32, lane);
    v16h bf3 = load_b_frag(Bs, 48, lane);
#pragma unroll
    for (int s = 0; s < 2; ++s) {
      v16h af = load_a_frag(As, mo + 16 * s, lane);
      acc[s][0] = __builtin_amdgcn_wmma_f32_16x16x32_f16(false, af, false, bf0, (short)0, acc[s][0], false, false);
      acc[s][1] = __builtin_amdgcn_wmma_f32_16x16x32_f16(false, af, false, bf1, (short)0, acc[s][1], false, false);
      acc[s][2] = __builtin_amdgcn_wmma_f32_16x16x32_f16(false, af, false, bf2, (short)0, acc[s][2], false, false);
      acc[s][3] = __builtin_amdgcn_wmma_f32_16x16x32_f16(false, af, false, bf3, (short)0, acc[s][3], false, false);
    }
    __syncthreads();
    aoff += 32 * a_k;
    woff += 32;
  }
#pragma unroll
  for (int s = 0; s < 2; ++s)
#pragma unroll
    for (int t = 0; t < 4; ++t) {
      int n = nbase + t * 16 + (lane & 15);
#pragma unroll
      for (int r = 0; r < 8; ++r) {
        int m = mbase + mo + 16 * s + (lo ? r : 8 + r);
        if (m < M) {
          float v = acc[s][t][r] + bias[n];
          if (relu && v < 0.f) v = 0.f;
          C[b * c_b + m * N + n] = v;
        }
      }
    }
}

// ---------------------------------------------------------------------------
// LayerNorm over 256-wide rows, in place. grid.x = rows, 256 threads.
// ---------------------------------------------------------------------------
__global__ void k_layernorm(float* __restrict__ X, const float* __restrict__ g,
                            const float* __restrict__ be) {
  __shared__ float red[256];
  long long row = blockIdx.x;
  int t = threadIdx.x;
  float x = X[row * 256 + t];
  red[t] = x;
  __syncthreads();
  for (int s = 128; s > 0; s >>= 1) { if (t < s) red[t] += red[t + s]; __syncthreads(); }
  float mu = red[0] * (1.f / 256.f);
  __syncthreads();
  float d = x - mu;
  red[t] = d * d;
  __syncthreads();
  for (int s = 128; s > 0; s >>= 1) { if (t < s) red[t] += red[t + s]; __syncthreads(); }
  float var = red[0] * (1.f / 256.f);
  X[row * 256 + t] = d * rsqrtf(var + 1e-5f) * g[t] + be[t];
}

// valid_W / valid_H from mask (downsampled mask: m[b,y,x] = mask[b,4y,4x])
__global__ void k_valid(const unsigned char* __restrict__ mask, float* __restrict__ vwh) {
  __shared__ int rH[128], rW[128];
  int b = blockIdx.x, t = threadIdx.x;
  rH[t] = mask[((long long)b * 512 + 4 * t) * 512] ? 0 : 1;
  rW[t] = mask[(long long)b * 512 * 512 + 4 * t] ? 0 : 1;
  __syncthreads();
  for (int s = 64; s > 0; s >>= 1) {
    if (t < s) { rH[t] += rH[t + s]; rW[t] += rW[t + s]; }
    __syncthreads();
  }
  if (t == 0) { vwh[b * 2] = (float)rW[0]; vwh[b * 2 + 1] = (float)rH[0]; }
}

// drop flags + output_proposals (logits or +inf)
__global__ void k_prop(const unsigned char* __restrict__ mask, const float* __restrict__ vwh,
                       unsigned char* __restrict__ drop, float* __restrict__ outprop) {
  int b = blockIdx.y;
  int n = blockIdx.x * 256 + threadIdx.x;
  int y = n >> 7, x = n & 127;
  unsigned char mv = mask[((long long)b * 512 + y * 4) * 512 + x * 4];
  float px = (x + 0.5f) / vwh[b * 2];
  float py = (y + 0.5f) / vwh[b * 2 + 1];
  bool valid = (px > 0.001f) && (px < 0.999f) && (py > 0.001f) && (py < 0.999f);
  bool dr = mv || !valid;
  long long i = (long long)b * HWF + n;
  drop[i] = dr ? 1 : 0;
  float inf = __builtin_inff();
  outprop[i * 2]     = dr ? inf : (logf(px) - log1pf(-px));
  outprop[i * 2 + 1] = dr ? inf : (logf(py) - log1pf(-py));
}

// cls head + scores = softmax(cls)[...,1] = sigmoid(c1-c0)
__global__ void k_cls(const float* __restrict__ om, const float* __restrict__ w,
                      const float* __restrict__ bi, float* __restrict__ out_cls,
                      float* __restrict__ scores) {
  long long i = (long long)blockIdx.x * 256 + threadIdx.x;
  const float* row = om + i * 256;
  float c0 = bi[0], c1 = bi[1];
  for (int k = 0; k < 256; ++k) {
    float v = row[k];
    c0 += v * w[k];
    c1 += v * w[256 + k];
  }
  out_cls[i * 2] = c0;
  out_cls[i * 2 + 1] = c1;
  scores[i] = 1.f / (1.f + expf(c0 - c1));
}

// coord head: coord_unact = h2@m3^T + b + outprop ; coord = flip(sigmoid)
__global__ void k_coord(const float* __restrict__ h2, const float* __restrict__ w,
                        const float* __restrict__ bi, const float* __restrict__ outprop,
                        float* __restrict__ coordun, float* __restrict__ out_coord) {
  long long i = (long long)blockIdx.x * 256 + threadIdx.x;
  const float* row = h2 + i * 256;
  float c0 = bi[0], c1 = bi[1];
  for (int k = 0; k < 256; ++k) {
    float v = row[k];
    c0 += v * w[k];
    c1 += v * w[256 + k];
  }
  c0 += outprop[i * 2];
  c1 += outprop[i * 2 + 1];
  coordun[i * 2] = c0;
  coordun[i * 2 + 1] = c1;
  out_coord[i * 2]     = 1.f / (1.f + expf(-c1));  // flip: (y,x)
  out_coord[i * 2 + 1] = 1.f / (1.f + expf(-c0));
}

// exact stable descending rank == lax.top_k order; scatter tc/ref by rank
__global__ void k_topk(const float* __restrict__ scores, const float* __restrict__ coordun,
                       float* __restrict__ tc, float* __restrict__ out_ref) {
  __shared__ float sh[256];
  int b = blockIdx.y;
  int i = blockIdx.x * 256 + threadIdx.x;
  const float* s = scores + (long long)b * HWF;
  float si = s[i];
  int cnt = 0;
  for (int jt = 0; jt < HWF / 256; ++jt) {
    sh[threadIdx.x] = s[jt * 256 + threadIdx.x];
    __syncthreads();
#pragma unroll 8
    for (int jj = 0; jj < 256; ++jj) {
      float sj = sh[jj];
      int j = jt * 256 + jj;
      cnt += (sj > si) || ((sj == si) && (j < i));
    }
    __syncthreads();
  }
  if (cnt < NQ) {
    long long src = ((long long)b * HWF + i) * 2;
    long long dst = ((long long)b * NQ + cnt) * 2;
    float c0 = coordun[src], c1 = coordun[src + 1];
    tc[dst] = c0;
    tc[dst + 1] = c1;
    out_ref[dst]     = 1.f / (1.f + expf(-c0));
    out_ref[dst + 1] = 1.f / (1.f + expf(-c1));
  }
}

// sine position embedding of tc -> (B,NQ,256)
__global__ void k_embed(const float* __restrict__ tc, float* __restrict__ emb) {
  long long idx = (long long)blockIdx.x * 256 + threadIdx.x;
  long long br = idx >> 8;
  int d = (int)(idx & 255);
  int c = d >> 7;
  int dd = d & 127;
  int j = dd >> 1;
  float t = tc[br * 2 + c];
  float pp = (1.f / (1.f + expf(-t))) * 6.283185307179586f;
  float v = pp * powf(10000.f, -(float)j * (1.f / 64.f));
  emb[idx] = (dd & 1) ? cosf(v) : sinf(v);
}

// bilinear grid-sample of mem at g = 2*ref-1 (zeros padding, align_corners=False)
__global__ void k_sample(const float* __restrict__ mem, const float* __restrict__ ref,
                         float* __restrict__ qs) {
  long long idx = (long long)blockIdx.x * 256 + threadIdx.x;
  long long br = idx >> 8;
  int c = (int)(idx & 255);
  int b = (int)(br / NQ);
  float x = 128.f * ref[br * 2]     - 0.5f;
  float y = 128.f * ref[br * 2 + 1] - 0.5f;
  float x0f = floorf(x), y0f = floorf(y);
  int x0 = (int)x0f, y0 = (int)y0f;
  float wx1 = x - x0f, wx0 = 1.f - wx1;
  float wy1 = y - y0f, wy0 = 1.f - wy1;
  const float* f = mem + ((long long)b * CDIM + c) * HWF;
  auto g = [&](int yy, int xx) -> float {
    if (yy < 0 || yy >= HF || xx < 0 || xx >= HF) return 0.f;
    return f[yy * HF + xx];
  };
  qs[idx] = g(y0, x0) * wx0 * wy0 + g(y0, x0 + 1) * wx1 * wy0 +
            g(y0 + 1, x0) * wx0 * wy1 + g(y0 + 1, x0 + 1) * wx1 * wy1;
}

// scatter qs -> query (B,C,H,W) and qpos -> query_pos (HW,B,C) at rounded pos
__global__ void k_scatter(const float* __restrict__ ref, const float* __restrict__ qs,
                          const float* __restrict__ qpos, float* __restrict__ out_query,
                          float* __restrict__ out_qp) {
  long long idx = (long long)blockIdx.x * 256 + threadIdx.x;
  long long br = idx >> 8;
  int c = (int)(idx & 255);
  int b = (int)(br / NQ);
  float rx = ref[br * 2], ry = ref[br * 2 + 1];
  int p = (int)(rintf(ry * 128.f) * 128.f + rintf(rx * 128.f));
  p = min(max(p, 0), HWF - 1);
  out_query[((long long)b * CDIM + c) * HWF + p] = qs[idx];
  out_qp[((long long)p * BATCH + b) * CDIM + c] = qpos[idx];
}

// ---------------------------------------------------------------------------
extern "C" void kernel_launch(void* const* d_in, const int* in_sizes, int n_in,
                              void* d_out, int out_size, void* d_ws, size_t ws_size,
                              hipStream_t stream) {
  const float* enc    = (const float*)d_in[0];
  const float* f4     = (const float*)d_in[1];
  const unsigned char* mask = (const unsigned char*)d_in[2];
  const float* conv_w = (const float*)d_in[3];
  const float* conv_b = (const float*)d_in[4];
  const float* mt_w   = (const float*)d_in[5];
  const float* mt_b   = (const float*)d_in[6];
  const float* mt_g   = (const float*)d_in[7];
  const float* mt_be  = (const float*)d_in[8];
  const float* ce_w   = (const float*)d_in[9];
  const float* ce_b   = (const float*)d_in[10];
  const float* m1_w   = (const float*)d_in[11];
  const float* m1_b   = (const float*)d_in[12];
  const float* m2_w   = (const float*)d_in[13];
  const float* m2_b   = (const float*)d_in[14];
  const float* m3_w   = (const float*)d_in[15];
  const float* m3_b   = (const float*)d_in[16];
  const float* pt_w   = (const float*)d_in[17];
  const float* pt_b   = (const float*)d_in[18];
  const float* pt_g   = (const float*)d_in[19];
  const float* pt_be  = (const float*)d_in[20];

  float* out = (float*)d_out;
  float* out_ref   = out;               // (8,14745,2)   = 235920
  float* out_query = out + 235920;      // (8,256,128,128)
  float* out_qp    = out + 33790352;    // (16384,8,256)
  float* out_cls   = out + 67344784;    // (8,16384,2)
  float* out_coord = out + 67606928;    // (8,16384,2)

  char* ws = (char*)d_ws;
  float* memb = (float*)(ws);                                // 128 MB: mem (B,256,HW)
  float* bufA = (float*)(ws + (size_t)128 * 1024 * 1024);    // 128 MB
  float* bufB = (float*)(ws + (size_t)256 * 1024 * 1024);    // 128 MB
  char* sm = ws + (size_t)384 * 1024 * 1024;
  float* outprop = (float*)sm;              sm += (size_t)BATCH * HWF * 2 * 4;
  float* coordun = (float*)sm;              sm += (size_t)BATCH * HWF * 2 * 4;
  float* scores  = (float*)sm;              sm += (size_t)BATCH * HWF * 4;
  float* tc      = (float*)sm;              sm += (size_t)BATCH * NQ * 2 * 4;
  float* vwh     = (float*)sm;              sm += 64;
  unsigned char* drop = (unsigned char*)sm;

  const int SBH = CDIM * HWF;  // batch stride of (B,HW,C)/(B,C,HW) (fits in int)
  const int SBQ = NQ * CDIM;

  // mask-derived proposals
  k_valid<<<BATCH, 128, 0, stream>>>(mask, vwh);
  k_prop<<<dim3(HWF / 256, BATCH), 256, 0, stream>>>(mask, vwh, drop, outprop);

  // mem = conv1x1(concat(up2x(enc), f4))
  k_gemm_mem<<<dim3(HWF / 64, CDIM / 128, BATCH), 128, 0, stream>>>(enc, f4, conv_w, conv_b, memb);

  // om_lin = masked(mem)^T @ mt_w^T + mt_b  -> bufA ; LayerNorm in place
  k_gemm<<<dim3(CDIM / 64, HWF / 128, BATCH), 128, 0, stream>>>(
      memb, SBH, 1, HWF, drop, HWF,
      mt_w, mt_b, bufA, SBH, HWF, CDIM, CDIM, 0);
  k_layernorm<<<BATCH * HWF, 256, 0, stream>>>(bufA, mt_g, mt_be);

  // cls + scores
  k_cls<<<BATCH * HWF / 256, 256, 0, stream>>>(bufA, ce_w, ce_b, out_cls, scores);

  // h1 = relu(om@m1^T+b) -> bufB ; h2 = relu(h1@m2^T+b) -> bufA
  k_gemm<<<dim3(CDIM / 64, HWF / 128, BATCH), 128, 0, stream>>>(
      bufA, SBH, CDIM, 1, nullptr, 0,
      m1_w, m1_b, bufB, SBH, HWF, CDIM, CDIM, 1);
  k_gemm<<<dim3(CDIM / 64, HWF / 128, BATCH), 128, 0, stream>>>(
      bufB, SBH, CDIM, 1, nullptr, 0,
      m2_w, m2_b, bufA, SBH, HWF, CDIM, CDIM, 1);

  // coord head + coord_unact
  k_coord<<<BATCH * HWF / 256, 256, 0, stream>>>(bufA, m3_w, m3_b, outprop, coordun, out_coord);

  // stable top-k by exact rank; writes tc + ref
  k_topk<<<dim3(HWF / 256, BATCH), 256, 0, stream>>>(scores, coordun, tc, out_ref);

  // pos-embed(tc) -> bufB ; qpos_lin = emb@pt_w^T+pt_b -> bufA ; LN in place
  k_embed<<<BATCH * NQ, 256, 0, stream>>>(tc, bufB);
  k_gemm<<<dim3(CDIM / 64, (NQ + 127) / 128, BATCH), 128, 0, stream>>>(
      bufB, SBQ, CDIM, 1, nullptr, 0,
      pt_w, pt_b, bufA, SBQ, NQ, CDIM, CDIM, 0);
  k_layernorm<<<BATCH * NQ, 256, 0, stream>>>(bufA, pt_g, pt_be);

  // grid-sample mem at ref -> qs in bufB
  k_sample<<<BATCH * NQ, 256, 0, stream>>>(memb, out_ref, bufB);

  // zero query + query_pos regions, then scatter
  hipMemsetAsync(out_query, 0, (size_t)2 * BATCH * CDIM * HWF * 4, stream);
  k_scatter<<<BATCH * NQ, 256, 0, stream>>>(out_ref, bufB, bufA, out_query, out_qp);
}
